// GraphLearner_38147899523080
// MI455X (gfx1250) — matrix operand order repaired and checked by reading, not compiled, not run on hardware
//
#include <hip/hip_runtime.h>
#include <hip/hip_bf16.h>

// Shapes (match reference constants)
#define BB   16
#define NN   1024
#define VF   2048
#define QF   1024
#define F1   3072          // VF+QF
#define MID  1024
#define TOPK 16
#define ROWS (BB*NN)       // 16384

#define KCHUNK 512
#define LDSTR  520         // padded LDS row stride in bf16 (1040B = 260 words == 4 mod 64 banks)

typedef __attribute__((ext_vector_type(16))) __bf16 v16bf;
typedef __attribute__((ext_vector_type(8)))  __bf16 v8bf;
typedef __attribute__((ext_vector_type(4)))  __bf16 v4bf;
typedef __attribute__((ext_vector_type(8)))  float  v8f;
typedef __attribute__((ext_vector_type(4)))  float  v4f;

__device__ __forceinline__ __bf16 f32_to_bf16(float f) {
    union { float f; unsigned int u; } v; v.f = f;
    unsigned int u = v.u + (0x7FFFu + ((v.u >> 16) & 1u));   // RNE
    unsigned short h = (unsigned short)(u >> 16);
    __bf16 r;
    __builtin_memcpy(&r, &h, 2);
    return r;
}

// Assemble a 16-bf16 A fragment (contiguous k = 16*half + s order, same as B) from two 16B LDS chunks.
__device__ __forceinline__ v16bf ld_afrag(const __bf16* sA, int row, int k0, int klo) {
    union { v16bf v; v8bf h[2]; } u;
    const __bf16* p = sA + row * LDSTR + k0 + 2 * klo;
    u.h[0] = *(const v8bf*)p;
    u.h[1] = *(const v8bf*)(p + 8);
    return u.v;
}

// ---------------- weight f32 -> bf16 conversion ----------------
__global__ __launch_bounds__(256) void cvt_bf16_kernel(const float* __restrict__ src,
                                                       __bf16* __restrict__ dst, int n) {
    int i = blockIdx.x * 256 + threadIdx.x;
    if (i < n) dst[i] = f32_to_bf16(src[i]);
}

// ---------------- GEMM1: h1 = relu([v|q] * W1^T + b1) ----------------
// 32 rows per block (2 row-tiles), 8 waves x 8 col-tiles = full MID.
// A panel staged chunk-wise in LDS (converted once), B streamed from L2.
__global__ __launch_bounds__(256) void gemm1_kernel(const float* __restrict__ v,
                                                    const float* __restrict__ q,
                                                    const __bf16* __restrict__ w1,
                                                    const float* __restrict__ b1,
                                                    __bf16* __restrict__ h1) {
    __shared__ __bf16 sA[32 * LDSTR];   // 33,280 B

    const int rowbase = blockIdx.x * 32;
    const int lane    = threadIdx.x & 31;
    const int wave    = threadIdx.x >> 5;
    const int halfsel = lane >> 4;
    const int lrow    = lane & 15;
    const int klo     = halfsel ? 8 : 0;

    v8f acc0[8], acc1[8];
#pragma unroll
    for (int n = 0; n < 8; ++n) {
        acc0[n] = (v8f){0.f,0.f,0.f,0.f,0.f,0.f,0.f,0.f};
        acc1[n] = (v8f){0.f,0.f,0.f,0.f,0.f,0.f,0.f,0.f};
    }

    for (int kc = 0; kc < F1 / KCHUNK; ++kc) {
        const int kbase = kc * KCHUNK;
        __syncthreads();
        // ---- stage 32 x 512 A chunk, f32 -> bf16 once per element ----
        for (int u = threadIdx.x; u < 32 * (KCHUNK / 4); u += 256) {
            const int row = u >> 7;               // /128 float4-units per row
            const int kq  = (u & 127) << 2;
            const int rg  = rowbase + row;
            const int k   = kbase + kq;
            v4f f;
            if (k < VF) f = *(const v4f*)(v + (size_t)rg * VF + k);
            else        f = *(const v4f*)(q + (size_t)(rg >> 10) * QF + (k - VF));
            v4bf b4;
#pragma unroll
            for (int i = 0; i < 4; ++i) b4[i] = f32_to_bf16(f[i]);
            *(v4bf*)(sA + row * LDSTR + kq) = b4;
        }
        __syncthreads();
        // ---- compute over this chunk ----
        for (int kk = 0; kk < KCHUNK / 32; ++kk) {
            const int k0 = kk * 32;
            v16bf a0 = ld_afrag(sA, lrow,      k0, klo);
            v16bf a1 = ld_afrag(sA, 16 + lrow, k0, klo);
            const int kg = kbase + k0;
#pragma unroll
            for (int n = 0; n < 8; ++n) {
                const int col = (wave * 8 + n) * 16 + lrow;
                v16bf bf = *(const v16bf*)(w1 + (size_t)col * F1 + kg + halfsel * 16);
                acc0[n] = __builtin_amdgcn_wmma_f32_16x16x32_bf16(false, a0, false, bf,
                                                                  (short)0, acc0[n], false, false);
                acc1[n] = __builtin_amdgcn_wmma_f32_16x16x32_bf16(false, a1, false, bf,
                                                                  (short)0, acc1[n], false, false);
            }
        }
    }

#pragma unroll
    for (int n = 0; n < 8; ++n) {
        const int col  = (wave * 8 + n) * 16 + lrow;
        const float bi = b1[col];
#pragma unroll
        for (int r = 0; r < 8; ++r) {
            const int m = r + halfsel * 8;
            float x0 = acc0[n][r] + bi; x0 = x0 > 0.f ? x0 : 0.f;
            float x1 = acc1[n][r] + bi; x1 = x1 > 0.f ? x1 : 0.f;
            h1[(size_t)(rowbase + m) * MID + col]      = f32_to_bf16(x0);
            h1[(size_t)(rowbase + 16 + m) * MID + col] = f32_to_bf16(x1);
        }
    }
}

// ---------------- GEMM2: h2 = relu(h1 * W2^T + b2) ----------------
__global__ __launch_bounds__(256) void gemm2_kernel(const __bf16* __restrict__ h1,
                                                    const __bf16* __restrict__ w2,
                                                    const float* __restrict__ b2,
                                                    __bf16* __restrict__ h2) {
    __shared__ __bf16 sA[32 * LDSTR];

    const int rowbase = blockIdx.x * 32;
    const int lane    = threadIdx.x & 31;
    const int wave    = threadIdx.x >> 5;
    const int halfsel = lane >> 4;
    const int lrow    = lane & 15;
    const int klo     = halfsel ? 8 : 0;

    v8f acc0[8], acc1[8];
#pragma unroll
    for (int n = 0; n < 8; ++n) {
        acc0[n] = (v8f){0.f,0.f,0.f,0.f,0.f,0.f,0.f,0.f};
        acc1[n] = (v8f){0.f,0.f,0.f,0.f,0.f,0.f,0.f,0.f};
    }

    for (int kc = 0; kc < MID / KCHUNK; ++kc) {
        const int kbase = kc * KCHUNK;
        __syncthreads();
        for (int u = threadIdx.x; u < 32 * (KCHUNK / 8); u += 256) {   // 16B units
            const int row = u >> 6;
            const int ku  = (u & 63) << 3;
            *(v8bf*)(sA + row * LDSTR + ku) =
                *(const v8bf*)(h1 + (size_t)(rowbase + row) * MID + kbase + ku);
        }
        __syncthreads();
        for (int kk = 0; kk < KCHUNK / 32; ++kk) {
            const int k0 = kk * 32;
            v16bf a0 = ld_afrag(sA, lrow,      k0, klo);
            v16bf a1 = ld_afrag(sA, 16 + lrow, k0, klo);
            const int kg = kbase + k0;
#pragma unroll
            for (int n = 0; n < 8; ++n) {
                const int col = (wave * 8 + n) * 16 + lrow;
                v16bf bf = *(const v16bf*)(w2 + (size_t)col * MID + kg + halfsel * 16);
                acc0[n] = __builtin_amdgcn_wmma_f32_16x16x32_bf16(false, a0, false, bf,
                                                                  (short)0, acc0[n], false, false);
                acc1[n] = __builtin_amdgcn_wmma_f32_16x16x32_bf16(false, a1, false, bf,
                                                                  (short)0, acc1[n], false, false);
            }
        }
    }

#pragma unroll
    for (int n = 0; n < 8; ++n) {
        const int col  = (wave * 8 + n) * 16 + lrow;
        const float bi = b2[col];
#pragma unroll
        for (int r = 0; r < 8; ++r) {
            const int m = r + halfsel * 8;
            float x0 = acc0[n][r] + bi; x0 = x0 > 0.f ? x0 : 0.f;
            float x1 = acc1[n][r] + bi; x1 = x1 > 0.f ? x1 : 0.f;
            h2[(size_t)(rowbase + m) * MID + col]      = f32_to_bf16(x0);
            h2[(size_t)(rowbase + 16 + m) * MID + col] = f32_to_bf16(x1);
        }
    }
}

// ---------------- GEMM3 (h2 * h2^T per batch) fused with top-K + softmax ----------------
__global__ __launch_bounds__(256) void gram_topk_kernel(const __bf16* __restrict__ h2,
                                                        float* __restrict__ out_adj,
                                                        float* __restrict__ out_idx) {
    __shared__ float  slog[16 * NN];     // 64 KB logit slab
    __shared__ __bf16 sA[16 * LDSTR];    // 16.6 KB staged A panel chunk

    const int rt      = blockIdx.x;          // 0..1023
    const int batch   = rt >> 6;
    const int lane    = threadIdx.x & 31;
    const int wave    = threadIdx.x >> 5;
    const int halfsel = lane >> 4;
    const int lrow    = lane & 15;
    const int klo     = halfsel ? 8 : 0;

    v8f acc[8];
#pragma unroll
    for (int n = 0; n < 8; ++n) acc[n] = (v8f){0.f,0.f,0.f,0.f,0.f,0.f,0.f,0.f};

    for (int kc = 0; kc < MID / KCHUNK; ++kc) {
        const int kbase = kc * KCHUNK;
        __syncthreads();
        for (int u = threadIdx.x; u < 16 * (KCHUNK / 8); u += 256) {
            const int row = u >> 6;
            const int ku  = (u & 63) << 3;
            *(v8bf*)(sA + row * LDSTR + ku) =
                *(const v8bf*)(h2 + (size_t)(rt * 16 + row) * MID + kbase + ku);
        }
        __syncthreads();
        for (int kk = 0; kk < KCHUNK / 32; ++kk) {
            const int k0 = kk * 32;
            v16bf a = ld_afrag(sA, lrow, k0, klo);
            const int kg = kbase + k0;
#pragma unroll
            for (int n = 0; n < 8; ++n) {
                const int ct = wave * 8 + n;
                const __bf16* bp = h2 + (size_t)(batch * NN + ct * 16 + lrow) * MID
                                      + kg + halfsel * 16;
                v16bf bf = *(const v16bf*)bp;
                acc[n] = __builtin_amdgcn_wmma_f32_16x16x32_bf16(false, a, false, bf,
                                                                 (short)0, acc[n], false, false);
            }
        }
    }

    // Dump logits to LDS
#pragma unroll
    for (int n = 0; n < 8; ++n) {
        const int col = (wave * 8 + n) * 16 + lrow;
#pragma unroll
        for (int r = 0; r < 8; ++r) {
            const int m = r + halfsel * 8;
            slog[m * NN + col] = acc[n][r];
        }
    }
    __syncthreads();

    // Per-row top-16 (iterative wave-wide argmax, lowest-index tie-break = jax top_k order)
    for (int rr = 0; rr < 2; ++rr) {
        const int m = wave * 2 + rr;
        float* rowp = slog + m * NN;
        float vals[TOPK]; int inds[TOPK];

        for (int t = 0; t < TOPK; ++t) {
            float bv = -3.4e38f; int bi = 0x7fffffff;
            for (int j = lane; j < NN; j += 32) {
                float x = rowp[j];
                if (x > bv || (x == bv && j < bi)) { bv = x; bi = j; }
            }
#pragma unroll
            for (int off = 16; off > 0; off >>= 1) {
                float ov = __shfl_xor(bv, off, 32);
                int   oi = __shfl_xor(bi, off, 32);
                if (ov > bv || (ov == bv && oi < bi)) { bv = ov; bi = oi; }
            }
            vals[t] = bv; inds[t] = bi;            // uniform after butterfly
            if (lane == 0) rowp[bi] = -3.4e38f;    // mask winner
            asm volatile("s_wait_dscnt 0" ::: "memory");
        }

        float mx = vals[0];
        float sum = 0.f;
#pragma unroll
        for (int t = 0; t < TOPK; ++t) sum = sum + expf(vals[t] - mx);

        if (lane < TOPK) {
            const int rg = rt * 16 + m;
            out_adj[(size_t)rg * TOPK + lane] = expf(vals[lane] - mx) / sum;
            out_idx[(size_t)rg * TOPK + lane] = (float)inds[lane];
        }
    }
}

extern "C" void kernel_launch(void* const* d_in, const int* in_sizes, int n_in,
                              void* d_out, int out_size, void* d_ws, size_t ws_size,
                              hipStream_t stream) {
    (void)in_sizes; (void)n_in; (void)out_size; (void)ws_size;
    const float* v  = (const float*)d_in[0];
    const float* q  = (const float*)d_in[1];
    // d_in[2] = v_mask (all ones, unused by reference math)
    // d_in[3] = top_K scalar (compile-time 16)
    const float* W1 = (const float*)d_in[4];
    const float* b1 = (const float*)d_in[5];
    const float* W2 = (const float*)d_in[6];
    const float* b2 = (const float*)d_in[7];

    char* ws = (char*)d_ws;
    __bf16* w1bf = (__bf16*)(ws);                    //  6 MB: 3072*1024 bf16
    __bf16* w2bf = (__bf16*)(ws + 6291456);          //  2 MB: 1024*1024 bf16
    __bf16* h1   = (__bf16*)(ws + 8388608);          // 32 MB: 16384*1024 bf16
    __bf16* h2   = (__bf16*)(ws + 41943040);         // 32 MB

    float* out_adj = (float*)d_out;                  // [16,1024,16]
    float* out_idx = out_adj + (size_t)ROWS * TOPK;  // [16,1024,16]

    cvt_bf16_kernel<<<(F1 * MID) / 256, 256, 0, stream>>>(W1, w1bf, F1 * MID);
    cvt_bf16_kernel<<<(MID * MID) / 256, 256, 0, stream>>>(W2, w2bf, MID * MID);
    gemm1_kernel<<<ROWS / 32, 256, 0, stream>>>(v, q, w1bf, b1, h1);
    gemm2_kernel<<<ROWS / 32, 256, 0, stream>>>(h1, w2bf, b2, h2);
    gram_topk_kernel<<<ROWS / 16, 256, 0, stream>>>(h2, out_adj, out_idx);
}
